// SPR_RGCN_88648124990836
// MI455X (gfx1250) — compile-verified, matrix operand order checked by reading, not run on hardware
//
#include <hip/hip_runtime.h>
#include <hip/hip_bf16.h>
#include <stdint.h>

#define Nn    50000
#define Ee    800000
#define Rr    3
#define Dd    128
#define Hh    128
#define Bb    256
#define NCLS  10
#define MAXP  25
#define EPSV  1e-5f

typedef __attribute__((ext_vector_type(16))) __bf16 v16bf;
typedef __attribute__((ext_vector_type(8)))  float  v8f;

// float -> bf16 (round to nearest even), manual so we never need scalar __bf16 math
__device__ __forceinline__ unsigned short f2bf(float f) {
    union { float f; uint32_t u; } x; x.f = f;
    uint32_t u = x.u;
    uint32_t r = u + 0x7FFFu + ((u >> 16) & 1u);
    if ((u & 0x7F800000u) == 0x7F800000u) r = u;   // NaN/Inf: truncate
    return (unsigned short)(r >> 16);
}

// ---------- node features: x = shape[i0] + color[i1] + pos[clip(i2)]  (bf16 out) ----------
__global__ void k_feat(const int* __restrict__ xi,
                       const float* __restrict__ se, const float* __restrict__ ce,
                       const float* __restrict__ pe, unsigned short* __restrict__ xbf) {
    int idx = blockIdx.x * blockDim.x + threadIdx.x;
    if (idx >= Nn * Hh) return;
    int n = idx >> 7, h = idx & 127;
    int i0 = xi[n * 3 + 0], i1 = xi[n * 3 + 1], i2 = xi[n * 3 + 2];
    i2 = i2 < 0 ? 0 : (i2 > MAXP - 1 ? MAXP - 1 : i2);
    float v = se[i0 * Dd + h] + ce[i1 * Dd + h] + pe[i2 * Dd + h];
    xbf[idx] = f2bf(v);
}

// ---------- pack weights: wt[slot][h][d] = bf16(W[slot][d][h]), slot 3 = root ----------
__global__ void k_pack(const float* __restrict__ W, const float* __restrict__ root,
                       unsigned short* __restrict__ wt) {
    int idx = blockIdx.x * blockDim.x + threadIdx.x;
    if (idx >= 4 * Dd * Hh) return;
    int s = idx >> 14;
    int rem = idx & 16383;
    int h = rem >> 7, d = rem & 127;
    float v = (s < 3) ? W[(size_t)s * Dd * Hh + d * Hh + h] : root[d * Hh + h];
    wt[(size_t)s * Dd * Hh + h * Dd + d] = f2bf(v);
}

// ---------- WMMA GEMM: one wave -> 16 rows x 128 cols of one weight slot ----------
// slot 0..2 -> xw[r][n][h], slot 3 -> xroot[n][h]
__global__ void __launch_bounds__(256) k_gemm(const unsigned short* __restrict__ xbf,
                                              const unsigned short* __restrict__ wt,
                                              float* __restrict__ xw,
                                              float* __restrict__ xroot) {
    const int lane  = threadIdx.x & 31;
    const int wid   = threadIdx.x >> 5;
    const int wg    = blockIdx.x * 8 + wid;
    const int slot  = wg & 3;
    const int mtile = wg >> 2;
    if (mtile >= Nn / 16) return;                 // wave-uniform: EXEC stays all-ones

    const int mloc  = lane & 15;
    const int khalf = (lane >> 4) << 4;           // 0 or 16
    const int nloc  = lane & 15;

    const unsigned short* arow  = xbf + (size_t)(mtile * 16 + mloc) * Dd;
    const unsigned short* wslot = wt + (size_t)slot * Dd * Hh;

    v8f acc[8];
#pragma unroll
    for (int nt = 0; nt < 8; ++nt) acc[nt] = (v8f){0.f,0.f,0.f,0.f,0.f,0.f,0.f,0.f};

#pragma unroll
    for (int ks = 0; ks < 4; ++ks) {
        const int k0 = ks * 32;
        v16bf a = *(const v16bf*)(arow + k0 + khalf);
#pragma unroll
        for (int nt = 0; nt < 8; ++nt) {
            v16bf b = *(const v16bf*)(wslot + (size_t)(nt * 16 + nloc) * Dd + k0 + khalf);
            acc[nt] = __builtin_amdgcn_wmma_f32_16x16x32_bf16(
                false, a, false, b, (short)0, acc[nt], false, false);
        }
    }

    float* dstbase = (slot < 3) ? (xw + (size_t)slot * Nn * Hh) : xroot;
    const int rbase = mtile * 16 + ((lane >> 4) << 3);   // +8 for upper half-lanes
#pragma unroll
    for (int nt = 0; nt < 8; ++nt) {
#pragma unroll
        for (int v = 0; v < 8; ++v) {
            dstbase[(size_t)(rbase + v) * Hh + nt * 16 + nloc] = acc[nt][v];
        }
    }
}

// ---------- edge scatter: 32 lanes per edge, float4 gather + 4 f32 atomics ----------
__global__ void k_scatter(const int* __restrict__ ei, const int* __restrict__ et,
                          const float* __restrict__ xw,
                          float* __restrict__ ssum, float* __restrict__ cnt) {
    int t = blockIdx.x * blockDim.x + threadIdx.x;
    int e = t >> 5;
    int lane = t & 31;
    if (e >= Ee) return;
    int src = ei[e];
    int dst = ei[Ee + e];
    int r   = et[e];
    const float4 v = *(const float4*)(xw + ((size_t)r * Nn + src) * Hh + lane * 4);
    float* sp = ssum + ((size_t)dst * Rr + r) * Hh + lane * 4;
    atomicAdd(sp + 0, v.x);
    atomicAdd(sp + 1, v.y);
    atomicAdd(sp + 2, v.z);
    atomicAdd(sp + 3, v.w);
    if (lane == 0) atomicAdd(&cnt[dst * Rr + r], 1.0f);
}

// ---------- per-relation mean + root + bias, BN(eval), ReLU -> bf16 + f32 ----------
__global__ void k_combine(const float* __restrict__ ssum, const float* __restrict__ cnt,
                          const float* __restrict__ xroot,
                          const float* __restrict__ bias, const float* __restrict__ g,
                          const float* __restrict__ beta, const float* __restrict__ mm,
                          const float* __restrict__ vv,
                          unsigned short* __restrict__ xbf_out, float* __restrict__ xf_out) {
    int idx = blockIdx.x * blockDim.x + threadIdx.x;
    if (idx >= Nn * Hh) return;
    int n = idx >> 7, h = idx & 127;
    float s = 0.f;
#pragma unroll
    for (int r = 0; r < Rr; ++r) {
        float c  = cnt[n * Rr + r];
        float sv = ssum[((size_t)n * Rr + r) * Hh + h];
        s += (c > 0.f) ? sv / c : 0.f;
    }
    s += xroot[idx] + bias[h];
    float y = g[h] * (s - mm[h]) * rsqrtf(vv[h] + EPSV) + beta[h];
    y = fmaxf(y, 0.f);
    xbf_out[idx] = f2bf(y);
    xf_out[idx]  = y;
}

// ---------- global mean pool (atomic) ----------
__global__ void k_pool(const float* __restrict__ xf, const int* __restrict__ batch,
                       float* __restrict__ hsum, float* __restrict__ hcnt) {
    int t = blockIdx.x * blockDim.x + threadIdx.x;
    int n = t >> 5;
    int lane = t & 31;
    if (n >= Nn) return;
    int b = batch[n];
    const float4 v = *(const float4*)(xf + (size_t)n * Hh + lane * 4);
    float* hp = hsum + (size_t)b * Hh + lane * 4;
    atomicAdd(hp + 0, v.x);
    atomicAdd(hp + 1, v.y);
    atomicAdd(hp + 2, v.z);
    atomicAdd(hp + 3, v.w);
    if (lane == 0) atomicAdd(&hcnt[b], 1.0f);
}

// ---------- classifier head: out[b][c] = (hsum[b]/cnt[b]) . clsW[:,c] + clsb[c] ----------
__global__ void k_cls(const float* __restrict__ hsum, const float* __restrict__ hcnt,
                      const float* __restrict__ clsW, const float* __restrict__ clsb,
                      float* __restrict__ out) {
    int t = blockIdx.x * blockDim.x + threadIdx.x;
    if (t >= Bb * NCLS) return;
    int b = t / NCLS, c = t % NCLS;
    float cn = fmaxf(hcnt[b], 1.0f);
    float acc = 0.f;
    for (int h = 0; h < Hh; ++h)
        acc += (hsum[b * Hh + h] / cn) * clsW[h * NCLS + c];
    out[t] = acc + clsb[c];
}

extern "C" void kernel_launch(void* const* d_in, const int* in_sizes, int n_in,
                              void* d_out, int out_size, void* d_ws, size_t ws_size,
                              hipStream_t stream) {
    (void)in_sizes; (void)n_in; (void)out_size; (void)ws_size;
    const int*   x_idx = (const int*)d_in[0];
    const int*   eidx  = (const int*)d_in[1];
    const int*   etype = (const int*)d_in[2];
    const int*   batch = (const int*)d_in[3];
    const float* se    = (const float*)d_in[4];
    const float* ce    = (const float*)d_in[5];
    const float* pe    = (const float*)d_in[6];
    const float* W1    = (const float*)d_in[7];
    const float* root1 = (const float*)d_in[8];
    const float* b1    = (const float*)d_in[9];
    const float* g1    = (const float*)d_in[10];
    const float* be1   = (const float*)d_in[11];
    const float* m1    = (const float*)d_in[12];
    const float* v1    = (const float*)d_in[13];
    const float* W2    = (const float*)d_in[14];
    const float* root2 = (const float*)d_in[15];
    const float* b2    = (const float*)d_in[16];
    const float* g2    = (const float*)d_in[17];
    const float* be2   = (const float*)d_in[18];
    const float* m2    = (const float*)d_in[19];
    const float* v2    = (const float*)d_in[20];
    const float* clsW  = (const float*)d_in[21];
    const float* clsb  = (const float*)d_in[22];
    float* out = (float*)d_out;

    // carve workspace (256-B aligned slices)
    char* p = (char*)d_ws;
    auto carve = [&](size_t bytes) -> char* {
        char* q = p; p += (bytes + 255) & ~(size_t)255; return q;
    };
    unsigned short* xbf   = (unsigned short*)carve((size_t)Nn * Hh * 2);
    unsigned short* wt    = (unsigned short*)carve((size_t)4 * Dd * Hh * 2);
    float*          xw    = (float*)carve((size_t)Rr * Nn * Hh * 4);
    float*          xroot = (float*)carve((size_t)Nn * Hh * 4);
    float*          ssum  = (float*)carve((size_t)Rr * Nn * Hh * 4);
    float*          cnt   = (float*)carve((size_t)Rr * Nn * 4);
    float*          xf    = (float*)carve((size_t)Nn * Hh * 4);
    float*          hsum  = (float*)carve((size_t)Bb * Hh * 4);
    float*          hcnt  = (float*)carve((size_t)Bb * 4);

    const int gFeat = (Nn * Hh + 255) / 256;
    const int gPack = (4 * Dd * Hh + 255) / 256;
    const int gGemm = ((Nn / 16) * 4 + 7) / 8;       // 8 waves per 256-thread block
    const int gScat = (Ee * 32 + 255) / 256;
    const int gPool = (Nn * 32 + 255) / 256;

    // ---- layer 1 ----
    k_feat<<<gFeat, 256, 0, stream>>>(x_idx, se, ce, pe, xbf);
    k_pack<<<gPack, 256, 0, stream>>>(W1, root1, wt);
    k_gemm<<<gGemm, 256, 0, stream>>>(xbf, wt, xw, xroot);
    hipMemsetAsync(ssum, 0, (size_t)Rr * Nn * Hh * 4, stream);
    hipMemsetAsync(cnt,  0, (size_t)Rr * Nn * 4, stream);
    k_scatter<<<gScat, 256, 0, stream>>>(eidx, etype, xw, ssum, cnt);
    k_combine<<<gFeat, 256, 0, stream>>>(ssum, cnt, xroot, b1, g1, be1, m1, v1, xbf, xf);

    // ---- layer 2 ----
    k_pack<<<gPack, 256, 0, stream>>>(W2, root2, wt);
    k_gemm<<<gGemm, 256, 0, stream>>>(xbf, wt, xw, xroot);
    hipMemsetAsync(ssum, 0, (size_t)Rr * Nn * Hh * 4, stream);
    hipMemsetAsync(cnt,  0, (size_t)Rr * Nn * 4, stream);
    k_scatter<<<gScat, 256, 0, stream>>>(eidx, etype, xw, ssum, cnt);
    k_combine<<<gFeat, 256, 0, stream>>>(ssum, cnt, xroot, b2, g2, be2, m2, v2, xbf, xf);

    // ---- pool + head ----
    hipMemsetAsync(hsum, 0, (size_t)Bb * Hh * 4, stream);
    hipMemsetAsync(hcnt, 0, (size_t)Bb * 4, stream);
    k_pool<<<gPool, 256, 0, stream>>>(xf, batch, hsum, hcnt);
    k_cls<<<(Bb * NCLS + 63) / 64, 64, 0, stream>>>(hsum, hcnt, clsW, clsb, out);
}
